// FBOnlyJointImageEyePositionLikelihood_26448408609546
// MI455X (gfx1250) — matrix operand order repaired and checked by reading, not compiled, not run on hardware
//
#include <hip/hip_runtime.h>
#include <hip/hip_bf16.h>
#include <math.h>

// ---------------------------------------------------------------------------
// Problem constants (from the reference)
// ---------------------------------------------------------------------------
#define HH 128
#define WW 128
#define NPIX (HH * WW)            // 16384
#define N_CELLS 700
#define N_CELLS_PAD 704           // 44 tiles of 16
#define N_BINS_FILTER 250
#define N_HIST 30
#define N_JIT 120
#define N_FRAMES (N_HIST + N_JIT) // 150
#define N_BINS 1545
#define N_VALID (N_BINS - N_BINS_FILTER + 1) // 1296 (conv output length)
#define N_LOSS (N_BINS - N_BINS_FILTER)      // 1295 (loss length, drops last)
#define N_G 32
#define N_SHIFT 64                // 8x8 (dy,dx) combinations
#define MAGIC_F ((float)(400.0 / 750.0))

#define KBLOCKS (NPIX / 32)       // 512 K-blocks of 32
#define STILES (N_SHIFT / 16)     // 4
#define NTILES (N_CELLS_PAD / 16) // 44
#define A_ELEMS ((size_t)KBLOCKS * STILES * 32 * 16)  // 1,048,576 shorts / plane
#define B_ELEMS ((size_t)KBLOCKS * NTILES * 32 * 16)  // 11,534,336 shorts / plane

typedef __attribute__((ext_vector_type(16))) __bf16 v16bf;
typedef __attribute__((ext_vector_type(16))) unsigned short v16us;
typedef __attribute__((ext_vector_type(8)))  float v8f;

// fp32 -> bf16(hi) + bf16(lo) split, RNE rounding.
__device__ __forceinline__ void bf16_split(float x, unsigned short& h, unsigned short& l) {
    unsigned u = __builtin_bit_cast(unsigned, x);
    unsigned r = (u + 0x7FFFu + ((u >> 16) & 1u)) & 0xFFFF0000u;
    h = (unsigned short)(r >> 16);
    float rem = x - __builtin_bit_cast(float, r);
    unsigned u2 = __builtin_bit_cast(unsigned, rem);
    l = (unsigned short)((u2 + 0x7FFFu + ((u2 >> 16) & 1u)) >> 16);
}

// ---------------------------------------------------------------------------
// Kernel 1: build the 64 shifted images (the only distinct jittered frames).
// shifted[s][r*W+c] = (r>=dy && c>=dx) ? img[(r-dy)*W + (c-dx)] : 0,  s=dy*8+dx
// ---------------------------------------------------------------------------
__global__ __launch_bounds__(256) void k_shift_images(
    const float* __restrict__ img, float* __restrict__ shifted) {
    const int s  = blockIdx.x;        // 0..63
    const int dy = s >> 3;
    const int dx = s & 7;
    float* dst = shifted + s * NPIX;
    for (int p = threadIdx.x; p < NPIX; p += 256) {
        const int r = p >> 7;
        const int c = p & 127;
        float v = 0.0f;
        if (r >= dy && c >= dx) v = img[(r - dy) * WW + (c - dx)];
        dst[p] = v;
    }
}

// ---------------------------------------------------------------------------
// Kernel 2a: pack A (shifted images) into WMMA fragment layout, bf16 hi/lo.
// 16-bit A (16x32): lane L holds M = L&15; elems e=0..15 -> K = kb*32 + (L<16?0:16) + e
// Plane index: (((kb*STILES + stile)*32 + lane)*16 + e)
// ---------------------------------------------------------------------------
__global__ __launch_bounds__(256) void k_pack_a(
    const float* __restrict__ shifted,             // [64][16384]
    unsigned short* __restrict__ Ah,
    unsigned short* __restrict__ Al) {
    const int idx = blockIdx.x * 256 + threadIdx.x;     // over KBLOCKS*STILES*32
    if (idx >= KBLOCKS * STILES * 32) return;
    const int lane  = idx & 31;
    const int stile = (idx >> 5) & (STILES - 1);
    const int kb    = idx >> 7;
    const int m     = stile * 16 + (lane & 15);
    const int koff  = kb * 32 + ((lane < 16) ? 0 : 16);
    const float* src = shifted + (size_t)m * NPIX + koff;

    v16us h, l;
#pragma unroll
    for (int e = 0; e < 8; ++e) {
        unsigned short hh, ll;
        bf16_split(src[e], hh, ll);        h[e] = hh;     l[e] = ll;
        bf16_split(src[8 + e], hh, ll);    h[8 + e] = hh; l[8 + e] = ll;
    }
    *(v16us*)(Ah + (size_t)idx * 16) = h;
    *(v16us*)(Al + (size_t)idx * 16) = l;
}

// ---------------------------------------------------------------------------
// Kernel 2b: pack B (spat_filters^T) into WMMA fragment layout, bf16 hi/lo.
// 16-bit B (32x16): lane L holds N = L&15; elems e=0..15 -> K = kb*32 + (L<16?0:16) + e
// Plane index: (((kb*NTILES + ntile)*32 + lane)*16 + e).  Cells >= 700 -> zeros.
// ---------------------------------------------------------------------------
__global__ __launch_bounds__(256) void k_pack_b(
    const float* __restrict__ spat,                // [700][16384]
    unsigned short* __restrict__ Bh,
    unsigned short* __restrict__ Bl) {
    const int idx = blockIdx.x * 256 + threadIdx.x;     // over KBLOCKS*NTILES*32
    if (idx >= KBLOCKS * NTILES * 32) return;
    const int lane  = idx & 31;
    const int ntile = (idx >> 5) % NTILES;
    const int kb    = idx / (NTILES * 32);
    const int c     = ntile * 16 + (lane & 15);
    const int koff  = kb * 32 + ((lane < 16) ? 0 : 16);

    v16us h = {}, l = {};
    if (c < N_CELLS) {
        const float* src = spat + (size_t)c * NPIX + koff;
#pragma unroll
        for (int e = 0; e < 16; ++e) {
            unsigned short hh, ll;
            bf16_split(src[e], hh, ll);
            h[e] = hh; l[e] = ll;
        }
    }
    *(v16us*)(Bh + (size_t)idx * 16) = h;
    *(v16us*)(Bl + (size_t)idx * 16) = l;
}

// ---------------------------------------------------------------------------
// Kernel 3: WMMA GEMM from packed fragments.
// shiftdot[64 x 704] = shifted(64 x 16384) * spat^T, bf16x3 ~= fp32 accuracy.
// One wave per 16x16 tile. Inner loop: 4x 32B fragment loads + 3 WMMAs.
// C/D layout: VGPR v, lane L -> row v + (L<16?0:8), col L&15.
// ---------------------------------------------------------------------------
__global__ __launch_bounds__(32) void k_gemm_shiftdot(
    const unsigned short* __restrict__ Ah, const unsigned short* __restrict__ Al,
    const unsigned short* __restrict__ Bh, const unsigned short* __restrict__ Bl,
    float* __restrict__ shiftdot) {            // [64][704]
    const int lane  = threadIdx.x;             // 0..31
    const int stile = blockIdx.x;              // 0..3
    const int ntile = blockIdx.y;              // 0..43

    size_t aIdx = (size_t)(stile * 32 + lane) * 16;
    size_t bIdx = (size_t)(ntile * 32 + lane) * 16;
    const size_t aStride = (size_t)STILES * 32 * 16;   // per K-block
    const size_t bStride = (size_t)NTILES * 32 * 16;

    v8f acc = {};
#pragma unroll 2
    for (int kb = 0; kb < KBLOCKS; ++kb) {
        const v16bf ahi = __builtin_bit_cast(v16bf, *(const v16us*)(Ah + aIdx));
        const v16bf alo = __builtin_bit_cast(v16bf, *(const v16us*)(Al + aIdx));
        const v16bf bhi = __builtin_bit_cast(v16bf, *(const v16us*)(Bh + bIdx));
        const v16bf blo = __builtin_bit_cast(v16bf, *(const v16us*)(Bl + bIdx));
        // D = A*B + C ; bf16x3: hi*hi + hi*lo + lo*hi (lo*lo ~ 2^-32, dropped)
        acc = __builtin_amdgcn_wmma_f32_16x16x32_bf16(false, ahi, false, bhi,
                                                      (short)0, acc, false, false);
        acc = __builtin_amdgcn_wmma_f32_16x16x32_bf16(false, ahi, false, blo,
                                                      (short)0, acc, false, false);
        acc = __builtin_amdgcn_wmma_f32_16x16x32_bf16(false, alo, false, bhi,
                                                      (short)0, acc, false, false);
        aIdx += aStride;
        bIdx += bStride;
    }

    const int col  = ntile * 16 + (lane & 15);
    const int rbase = stile * 16 + ((lane < 16) ? 0 : 8);
#pragma unroll
    for (int v = 0; v < 8; ++v)
        shiftdot[(rbase + v) * N_CELLS_PAD + col] = acc[v];
}

// ---------------------------------------------------------------------------
// Kernel 4: assemble spat_all[g][f][c]:
//   f < 30  -> history_frames_filt[f][c]  (broadcast over g)
//   f >= 30 -> shiftdot[dy*8+dx][c] with (dy,dx)=eye_movements[g][f-30]
// ---------------------------------------------------------------------------
__global__ __launch_bounds__(256) void k_fill_spat_all(
    const float* __restrict__ shiftdot,   // [64][704]
    const float* __restrict__ hist,       // [30][700]
    const int*   __restrict__ em,         // [32][120][2]
    float* __restrict__ spat_all) {       // [32][150][700]
    const int idx = blockIdx.x * 256 + threadIdx.x;
    const int total = N_G * N_FRAMES * N_CELLS;
    if (idx >= total) return;
    const int c  = idx % N_CELLS;
    const int gf = idx / N_CELLS;
    const int f  = gf % N_FRAMES;
    const int g  = gf / N_FRAMES;
    float v;
    if (f < N_HIST) {
        v = hist[f * N_CELLS + c];
    } else {
        const int j = f - N_HIST;
        int dy = em[(g * N_JIT + j) * 2 + 0];
        int dx = em[(g * N_JIT + j) * 2 + 1];
        dy = min(max(dy, 0), 7);
        dx = min(max(dx, 0), 7);
        v = shiftdot[(dy * 8 + dx) * N_CELLS_PAD + c];
    }
    spat_all[idx] = v;
}

// ---------------------------------------------------------------------------
// Kernel 5: fused upsample (frame->bin) + depthwise 250-tap conv + Poisson
// loss, one block per (cell, group). All operands staged in LDS (~8.8 KB of
// the 320 KB/WGP); deterministic tree reduction -> partial[g][c].
// ---------------------------------------------------------------------------
__global__ __launch_bounds__(256) void k_fused_loss(
    const float* __restrict__ spat_all,  // [32][150][700]
    const int*   __restrict__ fsel,      // [1545][2]
    const float* __restrict__ fwts,      // [1545][2]
    const float* __restrict__ tc,        // [700][250]
    const float* __restrict__ fb,        // [700][1296]
    const float* __restrict__ spikes,    // [700][1545]
    const float* __restrict__ tmask,     // [1295]
    float* __restrict__ partial) {       // [32][700]
    __shared__ float frames[N_FRAMES];
    __shared__ float ups[N_BINS];
    __shared__ float tcl[N_BINS_FILTER];
    __shared__ float red[256];

    const int c   = blockIdx.x;   // 0..699
    const int g   = blockIdx.y;   // 0..31
    const int tid = threadIdx.x;

    for (int f = tid; f < N_FRAMES; f += 256)
        frames[f] = spat_all[(g * N_FRAMES + f) * N_CELLS + c];
    for (int k = tid; k < N_BINS_FILTER; k += 256)
        tcl[k] = tc[c * N_BINS_FILTER + k];
    __syncthreads();

    for (int b = tid; b < N_BINS; b += 256) {
        const int   s0 = fsel[2 * b], s1 = fsel[2 * b + 1];
        const float w0 = fwts[2 * b], w1 = fwts[2 * b + 1];
        ups[b] = w0 * frames[s0] + w1 * frames[s1];
    }
    __syncthreads();

    float local = 0.0f;
    for (int t = tid; t < N_LOSS; t += 256) {
        float acc = 0.0f;
#pragma unroll 5
        for (int k = 0; k < N_BINS_FILTER; ++k)
            acc = __fmaf_rn(tcl[k], ups[t + k], acc);
        const float gen = acc + fb[c * N_VALID + t];
        const float sp  = spikes[c * N_BINS + N_BINS_FILTER + t];
        local += (expf(gen) - sp * gen) * tmask[t];
    }
    red[tid] = local;
    __syncthreads();
#pragma unroll
    for (int s = 128; s > 0; s >>= 1) {
        if (tid < s) red[tid] += red[tid + s];
        __syncthreads();
    }
    if (tid == 0) partial[g * N_CELLS + c] = red[0];
}

// ---------------------------------------------------------------------------
// Kernel 6: deterministic final reduce over cells; applies MAGIC scale.
// ---------------------------------------------------------------------------
__global__ __launch_bounds__(256) void k_final_reduce(
    const float* __restrict__ partial, float* __restrict__ out) {
    __shared__ float red[256];
    const int g   = blockIdx.x;
    const int tid = threadIdx.x;
    float local = 0.0f;
    for (int c = tid; c < N_CELLS; c += 256) local += partial[g * N_CELLS + c];
    red[tid] = local;
    __syncthreads();
#pragma unroll
    for (int s = 128; s > 0; s >>= 1) {
        if (tid < s) red[tid] += red[tid + s];
        __syncthreads();
    }
    if (tid == 0) out[g] = red[0] * MAGIC_F;
}

// ---------------------------------------------------------------------------
// Launcher
// ---------------------------------------------------------------------------
extern "C" void kernel_launch(void* const* d_in, const int* in_sizes, int n_in,
                              void* d_out, int out_size, void* d_ws, size_t ws_size,
                              hipStream_t stream) {
    const float* img    = (const float*)d_in[0];   // [128][128]
    const float* spikes = (const float*)d_in[1];   // [700][1545]
    const int*   em     = (const int*)  d_in[2];   // [32][120][2]
    const float* tmask  = (const float*)d_in[3];   // [1295]
    const float* spat   = (const float*)d_in[4];   // [700][16384]
    const float* tc     = (const float*)d_in[5];   // [700][250]
    const float* fb     = (const float*)d_in[6];   // [700][1296]
    const float* hist   = (const float*)d_in[7];   // [30][700]
    const int*   fsel   = (const int*)  d_in[8];   // [1545][2]
    const float* fwts   = (const float*)d_in[9];   // [1545][2]
    float* out = (float*)d_out;                    // [32]

    // Workspace carve-up. Float section first (64B-aligned total), then the
    // packed bf16 fragment planes.
    float* ws       = (float*)d_ws;
    float* shifted  = ws;                                   // 64*16384
    float* shiftdot = shifted  + (size_t)N_SHIFT * NPIX;    // 64*704
    float* spat_all = shiftdot + (size_t)N_SHIFT * N_CELLS_PAD; // 32*150*700
    float* partial  = spat_all + (size_t)N_G * N_FRAMES * N_CELLS; // 32*700
    float* fend     = partial  + (size_t)N_G * N_CELLS;
    unsigned short* Ah = (unsigned short*)fend;
    unsigned short* Al = Ah + A_ELEMS;
    unsigned short* Bh = Al + A_ELEMS;
    unsigned short* Bl = Bh + B_ELEMS;

    k_shift_images<<<N_SHIFT, 256, 0, stream>>>(img, shifted);

    const int aThreads = KBLOCKS * STILES * 32;   // 65,536
    k_pack_a<<<(aThreads + 255) / 256, 256, 0, stream>>>(shifted, Ah, Al);

    const int bThreads = KBLOCKS * NTILES * 32;   // 720,896
    k_pack_b<<<(bThreads + 255) / 256, 256, 0, stream>>>(spat, Bh, Bl);

    k_gemm_shiftdot<<<dim3(STILES, NTILES), 32, 0, stream>>>(Ah, Al, Bh, Bl, shiftdot);

    const int total = N_G * N_FRAMES * N_CELLS;
    k_fill_spat_all<<<(total + 255) / 256, 256, 0, stream>>>(shiftdot, hist, em, spat_all);

    k_fused_loss<<<dim3(N_CELLS, N_G), 256, 0, stream>>>(
        spat_all, fsel, fwts, tc, fb, spikes, tmask, partial);

    k_final_reduce<<<N_G, 256, 0, stream>>>(partial, out);
}
